// Net_78254304133683
// MI455X (gfx1250) — matrix operand order repaired and checked by reading, not compiled, not run on hardware
//
#include <hip/hip_runtime.h>
#include <hip/hip_bf16.h>

typedef __attribute__((ext_vector_type(16))) __bf16 v16bf;
typedef __attribute__((ext_vector_type(8)))  float  v8f;

#define DIMS   64
#define EPSBN  1e-5f

// ---------------------------------------------------------------------------
// WMMA helpers (v_wmma_f32_16x16x32_bf16; ISA 7.12.2 layouts, wave32)
// ---------------------------------------------------------------------------
__device__ __forceinline__ v8f wmma_bf16(v16bf a, v16bf b, v8f c) {
  return __builtin_amdgcn_wmma_f32_16x16x32_bf16(false, a, false, b, (short)0, c,
                                                 false, false);
}

// A fragment: 16x32 bf16 tile from f32 row-major X (ld = 64).
// Per-lane data is one run of 16 contiguous floats -> 4x float4 loads + cvt.
// a[j] (j=0..15) <-> K = k0 + 16*half + j
__device__ __forceinline__ v16bf load_a_frag_f32(const float* __restrict__ X,
                                                 int row0, int k0, int lane) {
  int m = lane & 15;
  int half = (lane >> 4) & 1;
  const float* rp = X + (size_t)(row0 + m) * DIMS + k0 + half * 16;
  float4 x0 = *(const float4*)(rp);
  float4 x1 = *(const float4*)(rp + 4);
  float4 x2 = *(const float4*)(rp + 8);
  float4 x3 = *(const float4*)(rp + 12);
  v16bf a;
  a[0]  = (__bf16)x0.x; a[1]  = (__bf16)x0.y; a[2]  = (__bf16)x0.z; a[3]  = (__bf16)x0.w;
  a[4]  = (__bf16)x1.x; a[5]  = (__bf16)x1.y; a[6]  = (__bf16)x1.z; a[7]  = (__bf16)x1.w;
  a[8]  = (__bf16)x2.x; a[9]  = (__bf16)x2.y; a[10] = (__bf16)x2.z; a[11] = (__bf16)x2.w;
  a[12] = (__bf16)x3.x; a[13] = (__bf16)x3.y; a[14] = (__bf16)x3.z; a[15] = (__bf16)x3.w;
  return a;
}

// B fragment: 32x16 bf16 (K x N) from weights stored [O,K] row-major (ld = 64).
// Lane-local data is 16 contiguous bf16 -> one 32-byte vector load.
// b[j] <-> K = k0 + 16*half + j, column n = lane&15
__device__ __forceinline__ v16bf load_b_frag(const __bf16* __restrict__ W,
                                             int o0, int k0, int lane) {
  int n = lane & 15;
  int half = (lane >> 4) & 1;
  return *(const v16bf*)(W + (size_t)(o0 + n) * DIMS + k0 + half * 16);
}

// ---------------------------------------------------------------------------
// Generic GEMM:  Y[n, 0:64] = X[n, 0:64] @ W^T + bias   (K = 64, O = 64)
// one wave per 16-row tile, 8 waves per block
// ---------------------------------------------------------------------------
template <bool HAS_BIAS>
__global__ void gemm64_kernel(const float* __restrict__ X,
                              const __bf16* __restrict__ W,
                              const float* __restrict__ bias,
                              float* __restrict__ Y, int ldy, int nrows) {
  int wave = blockIdx.x * (blockDim.x >> 5) + (threadIdx.x >> 5);
  int lane = threadIdx.x & 31;
  int row0 = wave * 16;
  if (row0 >= nrows) return;  // wave-uniform: EXEC all-1 for WMMA

  v16bf a0 = load_a_frag_f32(X, row0, 0, lane);
  v16bf a1 = load_a_frag_f32(X, row0, 32, lane);
  int n = lane & 15;
  int half = (lane >> 4) & 1;

#pragma unroll
  for (int o0 = 0; o0 < 64; o0 += 16) {
    v16bf b0 = load_b_frag(W, o0, 0, lane);
    v16bf b1 = load_b_frag(W, o0, 32, lane);
    v8f acc = {};
    acc = wmma_bf16(a0, b0, acc);
    acc = wmma_bf16(a1, b1, acc);
    float bv = HAS_BIAS ? bias[o0 + n] : 0.0f;
    float* yp = Y + (size_t)(row0 + half * 8) * ldy + o0 + n;
#pragma unroll
    for (int i = 0; i < 8; ++i) yp[(size_t)i * ldy] = acc[i] + bv;
  }
}

// ---------------------------------------------------------------------------
// Fused GRU cell: gi = swh@Wih^T + bih ; gh = h@Whh^T + bhh ; gates ; h in place
// blockDim = 64 (2 waves, each owns a 16-node tile); 48 KB LDS
// ---------------------------------------------------------------------------
__global__ __launch_bounds__(64) void gru_kernel(
    const float* __restrict__ swh, float* __restrict__ h,
    const __bf16* __restrict__ Wih, const __bf16* __restrict__ Whh,
    const float* __restrict__ bih, const float* __restrict__ bhh, int nrows) {
  __shared__ float gbuf[2][2][16][192];  // [wave][gi/gh][node][col]
  int wid  = threadIdx.x >> 5;
  int lane = threadIdx.x & 31;
  int row0 = (blockIdx.x * 2 + wid) * 16;
  bool active = (row0 < nrows);  // wave-uniform

  if (active) {
    v16bf ax0 = load_a_frag_f32(swh, row0, 0, lane);
    v16bf ax1 = load_a_frag_f32(swh, row0, 32, lane);
    v16bf ah0 = load_a_frag_f32(h,   row0, 0, lane);
    v16bf ah1 = load_a_frag_f32(h,   row0, 32, lane);
    int n = lane & 15;
    int half = (lane >> 4) & 1;
#pragma unroll
    for (int o0 = 0; o0 < 192; o0 += 16) {
      v16bf b0 = load_b_frag(Wih, o0, 0, lane);
      v16bf b1 = load_b_frag(Wih, o0, 32, lane);
      v8f acc = {};
      acc = wmma_bf16(ax0, b0, acc);
      acc = wmma_bf16(ax1, b1, acc);
      float bv = bih[o0 + n];
#pragma unroll
      for (int i = 0; i < 8; ++i)
        gbuf[wid][0][half * 8 + i][o0 + n] = acc[i] + bv;

      b0 = load_b_frag(Whh, o0, 0, lane);
      b1 = load_b_frag(Whh, o0, 32, lane);
      v8f acc2 = {};
      acc2 = wmma_bf16(ah0, b0, acc2);
      acc2 = wmma_bf16(ah1, b1, acc2);
      float bv2 = bhh[o0 + n];
#pragma unroll
      for (int i = 0; i < 8; ++i)
        gbuf[wid][1][half * 8 + i][o0 + n] = acc2[i] + bv2;
    }
  }
  __syncthreads();
  if (active) {
    for (int idx = lane; idx < 16 * DIMS; idx += 32) {
      int node = idx >> 6;
      int f = idx & 63;
      float ir = gbuf[wid][0][node][f],       hr = gbuf[wid][1][node][f];
      float iz = gbuf[wid][0][node][64 + f],  hz = gbuf[wid][1][node][64 + f];
      float in = gbuf[wid][0][node][128 + f], hn = gbuf[wid][1][node][128 + f];
      float r = 1.0f / (1.0f + __expf(-(ir + hr)));
      float z = 1.0f / (1.0f + __expf(-(iz + hz)));
      float nn = tanhf(in + r * hn);
      size_t gi = (size_t)(row0 + node) * DIMS + f;
      float hold = h[gi];
      h[gi] = (1.0f - z) * nn + z * hold;
    }
  }
}

// ---------------------------------------------------------------------------
// Edge gather-scale-scatter: msg = x3[src, rel] * norm -> atomicAdd into swh[dst]
// 1 thread per (edge, 2 features); x3/swh resident in 192MB L2
// ---------------------------------------------------------------------------
__global__ void edge_kernel(const float* __restrict__ x3, const int* __restrict__ src,
                            const int* __restrict__ dst, const int* __restrict__ rel,
                            const float* __restrict__ norm, float* __restrict__ swh,
                            int nE) {
  long long gid = (long long)blockIdx.x * blockDim.x + threadIdx.x;
  long long e = gid >> 5;
  if (e >= nE) return;
  int fp = ((int)gid & 31) * 2;
  int s = src[e], d = dst[e], r = rel[e];
  float nm = norm[e];
  const float* mp = x3 + (size_t)s * 192 + r * 64 + fp;
  float2 m = *(const float2*)mp;
  float* op = swh + (size_t)d * DIMS + fp;
  atomicAdd(op,     m.x * nm);
  atomicAdd(op + 1, m.y * nm);
}

// ---------------------------------------------------------------------------
// BatchNorm1d (training-mode batch stats, biased variance) in 3 stages
// stats[0:64]=sum  stats[64:128]=sumsq  stats[128:192]=scale  stats[192:256]=shift
// ---------------------------------------------------------------------------
__global__ void bn_stats_kernel(const float* __restrict__ Y, int nrows,
                                float* __restrict__ stats) {
  __shared__ float ssum[64], ssq[64];
  int t = threadIdx.x;
  if (t < 64) { ssum[t] = 0.0f; ssq[t] = 0.0f; }
  __syncthreads();
  int c = t & 63;
  int rstart = blockIdx.x * (blockDim.x >> 6) + (t >> 6);
  int rstep = gridDim.x * (blockDim.x >> 6);
  float s = 0.0f, q = 0.0f;
  for (int r = rstart; r < nrows; r += rstep) {
    float v = Y[(size_t)r * DIMS + c];
    s += v;
    q += v * v;
  }
  atomicAdd(&ssum[c], s);
  atomicAdd(&ssq[c], q);
  __syncthreads();
  if (t < 64) {
    atomicAdd(&stats[t], ssum[t]);
    atomicAdd(&stats[64 + t], ssq[t]);
  }
}

__global__ void bn_finalize_kernel(const float* __restrict__ stats,
                                   const float* __restrict__ gamma,
                                   const float* __restrict__ beta, float invn,
                                   float* __restrict__ ss) {
  int c = threadIdx.x;  // 64 threads
  float mu = stats[c] * invn;
  float var = stats[64 + c] * invn - mu * mu;
  float sc = gamma[c] * rsqrtf(var + EPSBN);
  ss[c] = sc;
  ss[64 + c] = beta[c] - mu * sc;
}

template <bool RELU>
__global__ void bn_apply_kernel(const float* __restrict__ X, const float* __restrict__ ss,
                                float* __restrict__ Y, int nelem) {
  int i = blockIdx.x * blockDim.x + threadIdx.x;
  if (i >= nelem) return;
  int c = i & 63;
  float v = X[i] * ss[c] + ss[64 + c];
  if (RELU) v = fmaxf(v, 0.0f);
  Y[i] = v;
}

// ---------------------------------------------------------------------------
// Head epilogue: out[n,o] = sum_k relu(bn(yh[n,k])) * W2[o,k] + b2[o]
// ---------------------------------------------------------------------------
__global__ void head_out_kernel(const float* __restrict__ yh, const float* __restrict__ ss,
                                const float* __restrict__ W2, const float* __restrict__ b2,
                                float* __restrict__ out, int nrows, int nout) {
  int gid = blockIdx.x * blockDim.x + threadIdx.x;
  int n = gid / nout;
  int o = gid - n * nout;
  if (n >= nrows) return;
  const float* yp = yh + (size_t)n * DIMS;
  const float* wp = W2 + o * DIMS;
  float acc = b2[o];
#pragma unroll 8
  for (int k = 0; k < DIMS; ++k) {
    float v = fmaxf(yp[k] * ss[k] + ss[64 + k], 0.0f);
    acc += v * wp[k];
  }
  out[(size_t)n * nout + o] = acc;
}

__global__ void cvt_bf16_kernel(const float* __restrict__ X, __bf16* __restrict__ Y, int n) {
  int i = blockIdx.x * blockDim.x + threadIdx.x;
  if (i < n) Y[i] = (__bf16)X[i];
}

// Transposing conversion for rel_weight: src [r][k][o] (K,O row-major)
// -> dst [r][o][k] ([O,K] row-major) so all B fragments load contiguously.
__global__ void cvt_bf16_t_kernel(const float* __restrict__ X, __bf16* __restrict__ Y,
                                  int nmat) {
  int i = blockIdx.x * blockDim.x + threadIdx.x;
  if (i >= nmat * DIMS * DIMS) return;
  int r = i >> 12;          // / 4096
  int rem = i & 4095;
  int o = rem >> 6;
  int k = rem & 63;
  Y[(size_t)r * 4096 + o * 64 + k] = (__bf16)X[(size_t)r * 4096 + k * 64 + o];
}

// ---------------------------------------------------------------------------
extern "C" void kernel_launch(void* const* d_in, const int* in_sizes, int n_in,
                              void* d_out, int out_size, void* d_ws, size_t ws_size,
                              hipStream_t stream) {
  const float* v        = (const float*)d_in[0];
  const int*   src      = (const int*)d_in[1];
  const int*   dst      = (const int*)d_in[2];
  const int*   rel      = (const int*)d_in[3];
  const float* norm     = (const float*)d_in[4];
  const float* emb_W    = (const float*)d_in[5];
  const float* emb_b    = (const float*)d_in[6];
  const float* emb_g    = (const float*)d_in[7];
  const float* emb_be   = (const float*)d_in[8];
  const float* rel_w    = (const float*)d_in[9];
  const float* gru_Wih  = (const float*)d_in[10];
  const float* gru_Whh  = (const float*)d_in[11];
  const float* gru_bih  = (const float*)d_in[12];
  const float* gru_bhh  = (const float*)d_in[13];
  const float* ker_g    = (const float*)d_in[14];
  const float* ker_be   = (const float*)d_in[15];
  const float* a_W1     = (const float*)d_in[16];
  const float* a_b1     = (const float*)d_in[17];
  const float* a_g      = (const float*)d_in[18];
  const float* a_be     = (const float*)d_in[19];
  const float* a_W2     = (const float*)d_in[20];
  const float* a_b2     = (const float*)d_in[21];
  const float* b_W1     = (const float*)d_in[22];
  const float* b_b1     = (const float*)d_in[23];
  const float* b_g      = (const float*)d_in[24];
  const float* b_be     = (const float*)d_in[25];
  const float* b_W2     = (const float*)d_in[26];
  const float* b_b2     = (const float*)d_in[27];

  const int N = in_sizes[0] / DIMS;   // 100000
  const int E = in_sizes[1];          // 1600000

  // workspace layout
  float* of0   = (float*)d_ws;                 // N*64   feature buffer (h0)
  float* hbuf  = of0  + (size_t)N * DIMS;      // N*64   hidden state
  float* swh   = hbuf + (size_t)N * DIMS;      // N*64   aggregated messages (+ temp)
  float* x3    = swh  + (size_t)N * DIMS;      // N*192  per-relation transforms
  float* stats = x3   + (size_t)N * 192;       // 256 f32
  __bf16* embWb = (__bf16*)(stats + 256);
  __bf16* relWb = embWb + 64 * 64;             // transposed to [O,K]
  __bf16* Wihb  = relWb + 3 * 64 * 64;
  __bf16* Whhb  = Wihb  + 192 * 64;
  __bf16* aW1b  = Whhb  + 192 * 64;
  __bf16* bW1b  = aW1b  + 64 * 64;
  float* ss = stats + 128;

  const int tiles = (N + 15) / 16;
  const int gemmBlocks = (tiles + 7) / 8;   // 8 waves/block

  // weight conversion to bf16 (rel_weight transposed to [O,K])
  cvt_bf16_kernel<<<(64 * 64 + 255) / 256, 256, 0, stream>>>(emb_W, embWb, 64 * 64);
  cvt_bf16_t_kernel<<<(3 * 64 * 64 + 255) / 256, 256, 0, stream>>>(rel_w, relWb, 3);
  cvt_bf16_kernel<<<(192 * 64 + 255) / 256, 256, 0, stream>>>(gru_Wih, Wihb, 192 * 64);
  cvt_bf16_kernel<<<(192 * 64 + 255) / 256, 256, 0, stream>>>(gru_Whh, Whhb, 192 * 64);
  cvt_bf16_kernel<<<(64 * 64 + 255) / 256, 256, 0, stream>>>(a_W1, aW1b, 64 * 64);
  cvt_bf16_kernel<<<(64 * 64 + 255) / 256, 256, 0, stream>>>(b_W1, bW1b, 64 * 64);

  // Embedding: Linear -> BN -> ReLU   (swh used as temp)
  gemm64_kernel<true><<<gemmBlocks, 256, 0, stream>>>(v, embWb, emb_b, swh, DIMS, N);
  hipMemsetAsync(stats, 0, 128 * sizeof(float), stream);
  bn_stats_kernel<<<256, 256, 0, stream>>>(swh, N, stats);
  bn_finalize_kernel<<<1, 64, 0, stream>>>(stats, emb_g, emb_be, 1.0f / N, ss);
  bn_apply_kernel<true><<<(N * DIMS + 255) / 256, 256, 0, stream>>>(swh, ss, of0, N * DIMS);

  hipMemsetAsync(hbuf, 0, (size_t)N * DIMS * sizeof(float), stream);

  const float* feat = of0;
  for (int layer = 0; layer < 2; ++layer) {
    // x3[n, r, :] = feat[n,:] @ rel_weight[r]
    for (int r = 0; r < 3; ++r) {
      gemm64_kernel<false><<<gemmBlocks, 256, 0, stream>>>(feat, relWb + r * 64 * 64,
                                                           nullptr, x3 + r * 64, 192, N);
    }
    hipMemsetAsync(swh, 0, (size_t)N * DIMS * sizeof(float), stream);
    {
      long long tot = (long long)E * 32;
      edge_kernel<<<(unsigned)((tot + 255) / 256), 256, 0, stream>>>(x3, src, dst, rel,
                                                                    norm, swh, E);
    }
    gru_kernel<<<(tiles + 1) / 2, 64, 0, stream>>>(swh, hbuf, Wihb, Whhb,
                                                   gru_bih, gru_bhh, N);
    feat = hbuf;
  }

  // hk = BN(h)  (no relu), stored at x3; ya / yb follow in the x3 region
  hipMemsetAsync(stats, 0, 128 * sizeof(float), stream);
  bn_stats_kernel<<<256, 256, 0, stream>>>(hbuf, N, stats);
  bn_finalize_kernel<<<1, 64, 0, stream>>>(stats, ker_g, ker_be, 1.0f / N, ss);
  float* hk = x3;
  float* ya = x3 + (size_t)N * DIMS;
  float* yb = x3 + (size_t)2 * N * DIMS;
  bn_apply_kernel<false><<<(N * DIMS + 255) / 256, 256, 0, stream>>>(hbuf, ss, hk, N * DIMS);

  float* xa = (float*)d_out;
  float* xb = xa + (size_t)N * 2;

  // head A: Linear -> BN -> ReLU -> Linear(64->2)
  gemm64_kernel<true><<<gemmBlocks, 256, 0, stream>>>(hk, aW1b, a_b1, ya, DIMS, N);
  hipMemsetAsync(stats, 0, 128 * sizeof(float), stream);
  bn_stats_kernel<<<256, 256, 0, stream>>>(ya, N, stats);
  bn_finalize_kernel<<<1, 64, 0, stream>>>(stats, a_g, a_be, 1.0f / N, ss);
  head_out_kernel<<<(N * 2 + 255) / 256, 256, 0, stream>>>(ya, ss, a_W2, a_b2, xa, N, 2);

  // head B: Linear -> BN -> ReLU -> Linear(64->21)
  gemm64_kernel<true><<<gemmBlocks, 256, 0, stream>>>(hk, bW1b, b_b1, yb, DIMS, N);
  hipMemsetAsync(stats, 0, 128 * sizeof(float), stream);
  bn_stats_kernel<<<256, 256, 0, stream>>>(yb, N, stats);
  bn_finalize_kernel<<<1, 64, 0, stream>>>(stats, b_g, b_be, 1.0f / N, ss);
  head_out_kernel<<<(N * 21 + 255) / 256, 256, 0, stream>>>(yb, ss, b_W2, b_b2, xb, N, 21);
}
